// InteractiveWindowedSelfAttention_86131274154361
// MI455X (gfx1250) — compile-verified, matrix operand order checked by reading, not run on hardware
//
#include <hip/hip_runtime.h>

// ---------------------------------------------------------------------------
// InteractiveWindowedSelfAttention for MI455X (gfx1250, wave32, WMMA)
//   x:(32,64,64,256) fp32 ; windows 8x8, HEADS=8, DK=DV=32
// Pipeline: pack x->f16, pack weights -> qkv GEMMs -> 3x3 conv (shift-GEMM)
//           -> windowed attention (+local fusion) -> output GEMM (+bias)
// Matrix math: v_wmma_f32_16x16x32_f16.
// Data movement: global_load_async_to_lds_b128 double-buffered tiles.
// ---------------------------------------------------------------------------

typedef __attribute__((ext_vector_type(16))) _Float16 v16h;
typedef __attribute__((ext_vector_type(8)))  _Float16 v8h;
typedef __attribute__((ext_vector_type(8)))  float    v8f;

union FragH { v16h v; unsigned u[8]; _Float16 h[16]; };
union Pack8 { uint4 u4; _Float16 h[8]; };

#define WMMA16(a, b, c) \
    __builtin_amdgcn_wmma_f32_16x16x32_f16(false, (a), false, (b), (short)0, (c), false, false)

#define IMG_B   32
#define CH      256
#define M_TOT   (IMG_B * 64 * 64)         // 131072
#define HEADS   8
#define SCALE_F 0.17677669529663687f      // 1/sqrt(32)

// --------------------------------------------------- CDNA5 async global->LDS
__device__ __forceinline__ unsigned lds_off(const void* p) {
    return (unsigned)(size_t)p;           // low 32 bits of generic ptr = LDS offset
}
__device__ __forceinline__ void async_b128(unsigned lds, const void* g) {
    asm volatile("global_load_async_to_lds_b128 %0, %1, off"
                 :: "v"(lds), "v"((unsigned long long)(size_t)g)
                 : "memory");
}
#define WAIT_ASYNC(N) asm volatile("s_wait_asynccnt %0" :: "n"(N) : "memory")

// ----------------------------------------------------------- fragment loads
// A/B-fragment (16x32 f16): lane L -> row=L%16 ; K pairs per ISA 7.12.2 layout
#define LOAD_AFRAG(frag, ARR, ROW, HALF)                                   \
    _Pragma("unroll")                                                      \
    for (int _v = 0; _v < 8; ++_v) {                                       \
        int _k = ((_v >> 2) << 4) + ((HALF) << 3) + ((_v & 3) << 1);       \
        (frag).u[_v] = *(const unsigned*)&ARR[(ROW)][_k];                  \
    }

// -------------------------------------------------------------- pack kernels
__global__ __launch_bounds__(256) void pack_x16(const float* __restrict__ X,
                                                _Float16* __restrict__ X16) {
    size_t i = ((size_t)blockIdx.x * 256 + threadIdx.x) * 8;
    const float4* s = (const float4*)(X + i);
    float4 a = s[0], b = s[1];
    float fv[8] = {a.x, a.y, a.z, a.w, b.x, b.y, b.z, b.w};
    Pack8 p;
#pragma unroll
    for (int v = 0; v < 8; ++v) p.h[v] = (_Float16)fv[v];
    *(uint4*)(X16 + i) = p.u4;
}

// 1x1 conv weight [K=256][N=256] fp32 -> transposed f16 [N][K]
__global__ __launch_bounds__(256) void pack_w_t(const float* __restrict__ W,
                                                _Float16* __restrict__ WT) {
    int t = blockIdx.x * 256 + threadIdx.x;          // t = n*256 + k
    int n = t >> 8, k = t & 255;
    WT[t] = (_Float16)W[k * 256 + n];
}

// 3x3 conv weight [3][3][K][N] fp32 -> f16 [tap][N][K]
__global__ __launch_bounds__(256) void pack_wl_t(const float* __restrict__ W,
                                                 _Float16* __restrict__ WT) {
    int t = blockIdx.x * 256 + threadIdx.x;          // 9*65536 total
    int tap = t >> 16;
    int r = t & 65535;
    int n = r >> 8, k = r & 255;
    WT[t] = (_Float16)W[tap * 65536 + k * 256 + n];
}

// --------------------------------------------------------------- f16 GEMM
// Out16[M][256] = f16( A16[M][256] * WT[N][K]^T ) ; K=256, async double-buffer
__global__ __launch_bounds__(256) void gemm_f16(const _Float16* __restrict__ A,
                                                const _Float16* __restrict__ WT,
                                                _Float16* __restrict__ Out) {
    __shared__ _Float16 As[2][128][40];
    __shared__ _Float16 Bs[2][128][40];
    const int tid = threadIdx.x, lane = tid & 31, wid = tid >> 5;
    const int half = lane >> 4, l16 = lane & 15;
    const int wm = (wid & 1) * 64, wn = (wid >> 1) * 32;
    const int mb = blockIdx.x * 128, nb = blockIdx.y * 128;
    const int arow = tid >> 1, apart = (tid & 1) * 16;

    const _Float16* ga = A  + (size_t)(mb + arow) * 256 + apart;
    const _Float16* gb = WT + (size_t)(nb + arow) * 256 + apart;

    auto issue = [&](int kk, int buf) {
        unsigned la = lds_off(&As[buf][arow][apart]);
        unsigned lb = lds_off(&Bs[buf][arow][apart]);
        async_b128(la,      ga + kk);
        async_b128(la + 16, ga + kk + 8);
        async_b128(lb,      gb + kk);
        async_b128(lb + 16, gb + kk + 8);
    };

    v8f acc[4][2] = {};
    issue(0, 0);
    for (int ks = 0; ks < 8; ++ks) {
        const int buf = ks & 1;
        const bool more = (ks + 1) < 8;
        if (more) { issue((ks + 1) * 32, buf ^ 1); WAIT_ASYNC(4); }
        else      { WAIT_ASYNC(0); }
        __syncthreads();

        FragH af[4], bf[2];
#pragma unroll
        for (int i = 0; i < 4; ++i) { int r = wm + i * 16 + l16; LOAD_AFRAG(af[i], As[buf], r, half); }
#pragma unroll
        for (int j = 0; j < 2; ++j) { int r = wn + j * 16 + l16; LOAD_AFRAG(bf[j], Bs[buf], r, half); }
#pragma unroll
        for (int i = 0; i < 4; ++i)
#pragma unroll
            for (int j = 0; j < 2; ++j)
                acc[i][j] = WMMA16(af[i].v, bf[j].v, acc[i][j]);
        __syncthreads();
    }

#pragma unroll
    for (int i = 0; i < 4; ++i)
#pragma unroll
        for (int j = 0; j < 2; ++j) {
            int n = nb + wn + j * 16 + l16;
#pragma unroll
            for (int r = 0; r < 8; ++r) {
                int m = mb + wm + i * 16 + half * 8 + r;
                Out[(size_t)m * 256 + n] = (_Float16)acc[i][j][r];
            }
        }
}

// ------------------------------------------------------- 3x3 conv shift-GEMM
// local16 = f16( sum_{tap} shift(V) * WL[tap] + bias ) ; K = 9*256
__global__ __launch_bounds__(256) void conv3x3_f16(const _Float16* __restrict__ V,
                                                   const _Float16* __restrict__ WLT,
                                                   const float* __restrict__ Bias,
                                                   _Float16* __restrict__ Out) {
    __shared__ _Float16 As[2][128][40];
    __shared__ _Float16 Bs[2][128][40];
    const int tid = threadIdx.x, lane = tid & 31, wid = tid >> 5;
    const int half = lane >> 4, l16 = lane & 15;
    const int wm = (wid & 1) * 64, wn = (wid >> 1) * 32;
    const int mb = blockIdx.x * 128, nb = blockIdx.y * 128;
    const int arow = tid >> 1, apart = (tid & 1) * 16;

    const int p = mb + arow;          // pixel index of this thread's A row
    const int b = p >> 12;
    const int y = (p >> 6) & 63;
    const int x = p & 63;

    auto issue = [&](int ksIdx, int buf) {
        const int tap = ksIdx >> 3;
        const int kk  = (ksIdx & 7) << 5;
        const int dy = tap / 3 - 1, dx = tap % 3 - 1;
        const int sy = y + dy, sx = x + dx;
        const bool valid = ((unsigned)sy < 64u) && ((unsigned)sx < 64u);
        unsigned la = lds_off(&As[buf][arow][apart]);
        if (valid) {
            const _Float16* g = V + ((size_t)((b * 64 + sy) * 64 + sx)) * 256 + kk + apart;
            async_b128(la,      g);
            async_b128(la + 16, g + 8);
        } else {
            uint4 z = {0u, 0u, 0u, 0u};
            *(uint4*)&As[buf][arow][apart]     = z;
            *(uint4*)&As[buf][arow][apart + 8] = z;
        }
        const _Float16* gw = WLT + (size_t)tap * 65536 + (size_t)(nb + arow) * 256 + kk + apart;
        unsigned lb = lds_off(&Bs[buf][arow][apart]);
        async_b128(lb,      gw);
        async_b128(lb + 16, gw + 8);
    };

    v8f acc[4][2] = {};
    issue(0, 0);
    for (int ks = 0; ks < 72; ++ks) {
        const int buf = ks & 1;
        const bool more = (ks + 1) < 72;
        // Edge waves may issue only 2 async ops per tile (A path zero-filled via
        // ds stores), so wait conservatively on <=2: in-order completion then
        // guarantees the current tile's copies have landed.
        if (more) { issue(ks + 1, buf ^ 1); WAIT_ASYNC(2); }
        else      { WAIT_ASYNC(0); }
        __syncthreads();

        FragH af[4], bf[2];
#pragma unroll
        for (int i = 0; i < 4; ++i) { int r = wm + i * 16 + l16; LOAD_AFRAG(af[i], As[buf], r, half); }
#pragma unroll
        for (int j = 0; j < 2; ++j) { int r = wn + j * 16 + l16; LOAD_AFRAG(bf[j], Bs[buf], r, half); }
#pragma unroll
        for (int i = 0; i < 4; ++i)
#pragma unroll
            for (int j = 0; j < 2; ++j)
                acc[i][j] = WMMA16(af[i].v, bf[j].v, acc[i][j]);
        __syncthreads();
    }

#pragma unroll
    for (int i = 0; i < 4; ++i)
#pragma unroll
        for (int j = 0; j < 2; ++j) {
            int n = nb + wn + j * 16 + l16;
            float bv = Bias[n];
#pragma unroll
            for (int r = 0; r < 8; ++r) {
                int m = mb + wm + i * 16 + half * 8 + r;
                Out[(size_t)m * 256 + n] = (_Float16)(acc[i][j][r] + bv);
            }
        }
}

// ----------------------------------------------------- windowed attention
// One wave per (window, head). 64 tokens, d=32. Epilogue fuses + local_out.
__global__ __launch_bounds__(32) void attn_win(const _Float16* __restrict__ Q,
                                               const _Float16* __restrict__ K,
                                               const _Float16* __restrict__ V,
                                               const _Float16* __restrict__ L,
                                               _Float16* __restrict__ O) {
    __shared__ _Float16 qs[64][32];
    __shared__ _Float16 ks[64][32];
    __shared__ _Float16 vs[64][32];
    __shared__ _Float16 ps[64][72];

    const int lane = threadIdx.x;
    const int half = lane >> 4, l16 = lane & 15;
    const int win = blockIdx.x >> 3;
    const int h   = blockIdx.x & 7;
    const int b   = win >> 6;
    const int wy  = (win >> 3) & 7;
    const int wx  = win & 7;
    const int y0  = wy * 8, x0 = wx * 8;

    // async-stage q,k,v tiles straight into LDS
#pragma unroll
    for (int tt = 0; tt < 2; ++tt) {
        int t = lane + tt * 32;
        int yy = y0 + (t >> 3), xx = x0 + (t & 7);
        size_t base = ((size_t)((b * 64 + yy) * 64 + xx)) * 256 + h * 32;
#pragma unroll
        for (int c = 0; c < 4; ++c) {
            async_b128(lds_off(&qs[t][c * 8]), Q + base + c * 8);
            async_b128(lds_off(&ks[t][c * 8]), K + base + c * 8);
            async_b128(lds_off(&vs[t][c * 8]), V + base + c * 8);
        }
    }
    WAIT_ASYNC(0);
    __syncthreads();

    // B operand of S = q*k^T: ks is [token][dim] which is the Bt layout
    FragH kf[4];
#pragma unroll
    for (int j = 0; j < 4; ++j) { int r = j * 16 + l16; LOAD_AFRAG(kf[j], ks, r, half); }

    for (int i = 0; i < 4; ++i) {
        FragH qf; { int r = i * 16 + l16; LOAD_AFRAG(qf, qs, r, half); }
        v8f s[4] = {};
#pragma unroll
        for (int j = 0; j < 4; ++j) s[j] = WMMA16(qf.v, kf[j].v, s[j]);

#pragma unroll
        for (int r = 0; r < 8; ++r) {
            float m = s[0][r];
#pragma unroll
            for (int j = 1; j < 4; ++j) m = fmaxf(m, s[j][r]);
#pragma unroll
            for (int msk = 1; msk < 16; msk <<= 1) m = fmaxf(m, __shfl_xor(m, msk, 32));
            float e[4], sum = 0.f;
#pragma unroll
            for (int j = 0; j < 4; ++j) { e[j] = __expf((s[j][r] - m) * SCALE_F); sum += e[j]; }
#pragma unroll
            for (int msk = 1; msk < 16; msk <<= 1) sum += __shfl_xor(sum, msk, 32);
            float inv = 1.0f / sum;
            int row = i * 16 + half * 8 + r;
#pragma unroll
            for (int j = 0; j < 4; ++j) ps[row][j * 16 + l16] = (_Float16)(e[j] * inv);
        }
    }
    __syncthreads();

    // O = P * V  (M=64, K=64, N=32)
    v8f o[4][2] = {};
#pragma unroll
    for (int kk = 0; kk < 2; ++kk) {
        FragH vf[2];
#pragma unroll
        for (int jn = 0; jn < 2; ++jn)
#pragma unroll
            for (int v = 0; v < 8; ++v) {
                int kl = kk * 32 + ((v >> 2) << 4) + (half << 3) + ((v & 3) << 1);
                vf[jn].h[2 * v]     = vs[kl][jn * 16 + l16];
                vf[jn].h[2 * v + 1] = vs[kl + 1][jn * 16 + l16];
            }
#pragma unroll
        for (int i = 0; i < 4; ++i) {
            FragH pf;
#pragma unroll
            for (int v = 0; v < 8; ++v) {
                int kl = kk * 32 + ((v >> 2) << 4) + (half << 3) + ((v & 3) << 1);
                pf.u[v] = *(const unsigned*)&ps[i * 16 + l16][kl];
            }
#pragma unroll
            for (int jn = 0; jn < 2; ++jn) o[i][jn] = WMMA16(pf.v, vf[jn].v, o[i][jn]);
        }
    }

    // store sum16 = attention output + local branch (f16, NHWC)
#pragma unroll
    for (int i = 0; i < 4; ++i)
#pragma unroll
        for (int jn = 0; jn < 2; ++jn)
#pragma unroll
            for (int r = 0; r < 8; ++r) {
                int t = i * 16 + half * 8 + r;
                int yy = y0 + (t >> 3), xx = x0 + (t & 7);
                int ch = h * 32 + jn * 16 + l16;
                size_t idx = ((size_t)((b * 64 + yy) * 64 + xx)) * 256 + ch;
                O[idx] = (_Float16)(o[i][jn][r] + (float)L[idx]);
            }
}

// --------------------------------------------------------- output projection
// out_f32 = sum16 * WoutT^T + b_out ; async double-buffer
__global__ __launch_bounds__(256) void gemm_out_f32(const _Float16* __restrict__ A,
                                                    const _Float16* __restrict__ WT,
                                                    const float* __restrict__ Bias,
                                                    float* __restrict__ Out) {
    __shared__ _Float16 As[2][128][40];
    __shared__ _Float16 Bs[2][128][40];
    const int tid = threadIdx.x, lane = tid & 31, wid = tid >> 5;
    const int half = lane >> 4, l16 = lane & 15;
    const int wm = (wid & 1) * 64, wn = (wid >> 1) * 32;
    const int mb = blockIdx.x * 128, nb = blockIdx.y * 128;
    const int arow = tid >> 1, apart = (tid & 1) * 16;

    const _Float16* ga = A  + (size_t)(mb + arow) * 256 + apart;
    const _Float16* gb = WT + (size_t)(nb + arow) * 256 + apart;

    auto issue = [&](int kk, int buf) {
        unsigned la = lds_off(&As[buf][arow][apart]);
        unsigned lb = lds_off(&Bs[buf][arow][apart]);
        async_b128(la,      ga + kk);
        async_b128(la + 16, ga + kk + 8);
        async_b128(lb,      gb + kk);
        async_b128(lb + 16, gb + kk + 8);
    };

    v8f acc[4][2] = {};
    issue(0, 0);
    for (int ks = 0; ks < 8; ++ks) {
        const int buf = ks & 1;
        const bool more = (ks + 1) < 8;
        if (more) { issue((ks + 1) * 32, buf ^ 1); WAIT_ASYNC(4); }
        else      { WAIT_ASYNC(0); }
        __syncthreads();

        FragH af[4], bf[2];
#pragma unroll
        for (int i = 0; i < 4; ++i) { int r = wm + i * 16 + l16; LOAD_AFRAG(af[i], As[buf], r, half); }
#pragma unroll
        for (int j = 0; j < 2; ++j) { int r = wn + j * 16 + l16; LOAD_AFRAG(bf[j], Bs[buf], r, half); }
#pragma unroll
        for (int i = 0; i < 4; ++i)
#pragma unroll
            for (int j = 0; j < 2; ++j)
                acc[i][j] = WMMA16(af[i].v, bf[j].v, acc[i][j]);
        __syncthreads();
    }

#pragma unroll
    for (int i = 0; i < 4; ++i)
#pragma unroll
        for (int j = 0; j < 2; ++j) {
            int n = nb + wn + j * 16 + l16;
            float bv = Bias[n];
#pragma unroll
            for (int r = 0; r < 8; ++r) {
                int m = mb + wm + i * 16 + half * 8 + r;
                Out[(size_t)m * 256 + n] = acc[i][j][r] + bv;
            }
        }
}

// ---------------------------------------------------------------------------
extern "C" void kernel_launch(void* const* d_in, const int* in_sizes, int n_in,
                              void* d_out, int out_size, void* d_ws, size_t ws_size,
                              hipStream_t stream) {
    const float* x       = (const float*)d_in[0];
    const float* w_q     = (const float*)d_in[1];
    const float* w_k     = (const float*)d_in[2];
    const float* w_v     = (const float*)d_in[3];
    const float* w_local = (const float*)d_in[4];
    const float* b_local = (const float*)d_in[5];
    const float* w_out   = (const float*)d_in[6];
    const float* b_out   = (const float*)d_in[7];

    char* ws = (char*)d_ws;
    const size_t QB = (size_t)M_TOT * CH * sizeof(_Float16);   // 64 MiB
    _Float16* x16   = (_Float16*)(ws + 0 * QB);
    _Float16* q16   = (_Float16*)(ws + 1 * QB);
    _Float16* k16   = (_Float16*)(ws + 2 * QB);
    _Float16* v16   = (_Float16*)(ws + 3 * QB);
    _Float16* loc16 = (_Float16*)(ws + 4 * QB);
    _Float16* sum16 = (_Float16*)(ws + 5 * QB);
    _Float16* wqT   = (_Float16*)(ws + 6 * QB);
    _Float16* wkT   = wqT + 65536;
    _Float16* wvT   = wkT + 65536;
    _Float16* woT   = wvT + 65536;
    _Float16* wlT   = woT + 65536;
    const size_t need = 6 * QB + (size_t)13 * 65536 * sizeof(_Float16);
    if (ws_size < need) return;

    pack_x16<<<M_TOT * CH / (256 * 8), 256, 0, stream>>>(x, x16);
    pack_w_t<<<256, 256, 0, stream>>>(w_q, wqT);
    pack_w_t<<<256, 256, 0, stream>>>(w_k, wkT);
    pack_w_t<<<256, 256, 0, stream>>>(w_v, wvT);
    pack_w_t<<<256, 256, 0, stream>>>(w_out, woT);
    pack_wl_t<<<2304, 256, 0, stream>>>(w_local, wlT);

    dim3 g(M_TOT / 128, CH / 128);
    gemm_f16<<<g, 256, 0, stream>>>(x16, wqT, q16);
    gemm_f16<<<g, 256, 0, stream>>>(x16, wkT, k16);
    gemm_f16<<<g, 256, 0, stream>>>(x16, wvT, v16);

    conv3x3_f16<<<g, 256, 0, stream>>>(v16, wlT, b_local, loc16);

    attn_win<<<2048 * HEADS, 32, 0, stream>>>(q16, k16, v16, loc16, sum16);

    gemm_out_f32<<<g, 256, 0, stream>>>(sum16, woT, b_out, (float*)d_out);
}
